// DynKQuantizer2_33389075759173
// MI455X (gfx1250) — matrix operand order repaired and checked by reading, not compiled, not run on hardware
//
#include <hip/hip_runtime.h>
#include <hip/hip_bf16.h>

typedef __bf16 bf16;
typedef __attribute__((ext_vector_type(16))) __bf16 v16bf;
typedef __attribute__((ext_vector_type(8)))  __bf16 v8bf;
typedef __attribute__((ext_vector_type(8)))  float  v8f;

// ---------------------------------------------------------------------------
// CDNA5 async global->LDS staging (ASYNCcnt path, §15.18 opcodes 95-102).
// Each active lane moves 16 bytes to its own LDS byte offset.
// ---------------------------------------------------------------------------
__device__ __forceinline__ void async_b128(unsigned lds_byte_off, const bf16* g) {
    asm volatile("global_load_async_to_lds_b128 %0, %1, off"
                 :: "v"(lds_byte_off), "v"(g) : "memory");
}
__device__ __forceinline__ void wait_async0() {
    asm volatile("s_wait_asynccnt 0x0" ::: "memory");
}
// Generic pointer to a __shared__ object: low 32 bits == LDS byte address.
#define LDS_OFF(p) ((unsigned)(uintptr_t)(p))

// ---------------------------------------------------------------------------
// WMMA fragment loaders (CDNA5 wave32 layouts, 16-bit data, ld = 32 in LDS)
// A 16x32 (MxK): lane L: m=L%16, h=L/16; halves 0-7 -> K=8h+0..7,
//                halves 8-15 -> K=16+8h+0..7 (two 16B chunks per lane)
// B 32x16 (KxN): lane L: n=L%16, h=L/16; halves 0-7 -> K=8h+0..7,
//                halves 8-15 -> K=16+8h+0..7
// ---------------------------------------------------------------------------
__device__ __forceinline__ v16bf frag_a(const bf16* base) {
    const int L = threadIdx.x & 31, m = L & 15, h = L >> 4;
    union { v16bf v; v8bf p[2]; } u;
    const bf16* r = base + m * 32;
    u.p[0] = *(const v8bf*)(r + 8 * h);
    u.p[1] = *(const v8bf*)(r + 16 + 8 * h);
    return u.v;
}
__device__ __forceinline__ v16bf frag_b(const bf16* base) {
    const int L = threadIdx.x & 31, n = L & 15, h = L >> 4;
    union { v16bf v; v8bf p[2]; } u;
    const bf16* r = base + n * 32 + 8 * h;
    u.p[0] = *(const v8bf*)(r);
    u.p[1] = *(const v8bf*)(r + 16);
    return u.v;
}
__device__ __forceinline__ v8f wmma_bf16(v16bf a, v16bf b, v8f c) {
    return __builtin_amdgcn_wmma_f32_16x16x32_bf16(false, a, false, b,
                                                   (short)0, c, false, false);
}

// ---------------------------------------------------------------------------
// Tiled GEMM: C[M,N] = act(A[M,K] @ Wt[N,K]^T), A/Wt bf16 K-contiguous.
// 128x128 C tile / 256-thread block; wave grid 4x2, each wave 32x64
// (2 A-frags x 4 B-frags = 8 accumulators); K-step 32, double-buffered LDS
// filled via global_load_async_to_lds_b128.
// ---------------------------------------------------------------------------
#define GEMM_STEP(BUF, KB)                                                     \
    do {                                                                       \
        wait_async0();                                                         \
        __syncthreads();                                                       \
        if ((KB) + 32 < K) {                                                   \
            async_b128(lA0[(BUF) ^ 1], ga0 + (KB) + 32);                       \
            async_b128(lA1[(BUF) ^ 1], ga1 + (KB) + 32);                       \
            async_b128(lB0[(BUF) ^ 1], gb0 + (KB) + 32);                       \
            async_b128(lB1[(BUF) ^ 1], gb1 + (KB) + 32);                       \
            __builtin_prefetch(ga0 + (KB) + 64, 0, 1);                         \
        }                                                                      \
        {                                                                      \
            const bf16* Ac = &As[BUF][0];                                      \
            const bf16* Bc = &Bs[BUF][0];                                      \
            v16bf a0 = frag_a(Ac + (wr * 32) * 32);                            \
            v16bf a1 = frag_a(Ac + (wr * 32 + 16) * 32);                       \
            _Pragma("unroll") for (int j = 0; j < 4; ++j) {                    \
                v16bf bj = frag_b(Bc + (wc * 64 + j * 16) * 32);               \
                acc0[j] = wmma_bf16(a0, bj, acc0[j]);                          \
                acc1[j] = wmma_bf16(a1, bj, acc1[j]);                          \
            }                                                                  \
        }                                                                      \
    } while (0)

template <bool RELU, bool OUT_F32>
__global__ __launch_bounds__(256) void gemm_wmma(
    const bf16* __restrict__ A, const bf16* __restrict__ Wt,
    void* __restrict__ Cout, int N, int K)
{
    __shared__ __align__(16) bf16 As[2][128 * 32];   // 2 x 8 KB
    __shared__ __align__(16) bf16 Bs[2][128 * 32];   // 2 x 8 KB

    const int bx = blockIdx.x, by = blockIdx.y;
    const int t  = threadIdx.x;
    const int w  = t >> 5, wr = w & 3, wc = w >> 2;  // wave grid 4 x 2

    // Staging geometry: tile = 128 rows x 32 cols = 512 x 16B chunks;
    // thread t owns chunks t and t+256 (row = c>>2, col8 = (c&3)*8).
    const int c0 = t, c1 = t + 256;
    const int r0 = c0 >> 2, o0 = (c0 & 3) * 8;
    const int r1 = c1 >> 2, o1 = (c1 & 3) * 8;

    const bf16* Ab = A  + (size_t)bx * 128 * K;
    const bf16* Wb = Wt + (size_t)by * 128 * K;
    const bf16* ga0 = Ab + (size_t)r0 * K + o0;
    const bf16* ga1 = Ab + (size_t)r1 * K + o1;
    const bf16* gb0 = Wb + (size_t)r0 * K + o0;
    const bf16* gb1 = Wb + (size_t)r1 * K + o1;

    unsigned lA0[2], lA1[2], lB0[2], lB1[2];
#pragma unroll
    for (int s = 0; s < 2; ++s) {
        lA0[s] = LDS_OFF(&As[s][r0 * 32 + o0]);
        lA1[s] = LDS_OFF(&As[s][r1 * 32 + o1]);
        lB0[s] = LDS_OFF(&Bs[s][r0 * 32 + o0]);
        lB1[s] = LDS_OFF(&Bs[s][r1 * 32 + o1]);
    }

    v8f acc0[4] = {}, acc1[4] = {};

    // Prologue: fill buffer 0.
    async_b128(lA0[0], ga0); async_b128(lA1[0], ga1);
    async_b128(lB0[0], gb0); async_b128(lB1[0], gb1);

    for (int kb = 0; kb < K; kb += 64) {   // K is a multiple of 64
        GEMM_STEP(0, kb);
        GEMM_STEP(1, kb + 32);
    }

    // D layout: VGPR r, lanes 0-15 -> M=r, lanes 16-31 -> M=r+8; N = lane%16
    const int L = t & 31, nl = L & 15, h = L >> 4;
#pragma unroll
    for (int i = 0; i < 2; ++i) {
#pragma unroll
        for (int j = 0; j < 4; ++j) {
            const int row0 = bx * 128 + wr * 32 + i * 16 + 8 * h;
            const int col  = by * 128 + wc * 64 + j * 16 + nl;
#pragma unroll
            for (int r = 0; r < 8; ++r) {
                float v = i ? acc1[j][r] : acc0[j][r];
                if (RELU) v = fmaxf(v, 0.0f);
                const size_t o = (size_t)(row0 + r) * N + col;
                if (OUT_F32) ((float*)Cout)[o] = v;
                else         ((bf16*)Cout)[o]  = (bf16)v;
            }
        }
    }
}

// ---------------------------------------------------------------------------
__global__ void cvt_f32_to_bf16(const float* __restrict__ in,
                                bf16* __restrict__ out, int n) {
    int i = blockIdx.x * blockDim.x + threadIdx.x;
    const int stride = gridDim.x * blockDim.x;
    for (; i < n; i += stride) out[i] = (bf16)in[i];
}

// One wave32 per row: k[row] = argmax(scores[row]) + 1 (first-max tie break)
__global__ __launch_bounds__(256) void argmax_k(const float* __restrict__ S,
                                                int* __restrict__ kout, int ncols) {
    const int row  = blockIdx.x * 8 + (threadIdx.x >> 5);
    const int lane = threadIdx.x & 31;
    const float* r = S + (size_t)row * ncols;
    float best = -__builtin_huge_valf();
    int   bi   = ncols;
    for (int j = lane; j < ncols; j += 32) {
        float v = r[j];
        if (v > best) { best = v; bi = j; }
    }
#pragma unroll
    for (int off = 16; off > 0; off >>= 1) {
        float ov = __shfl_down(best, off, 32);
        int   oi = __shfl_down(bi, off, 32);
        if (ov > best || (ov == best && oi < bi)) { best = ov; bi = oi; }
    }
    if (lane == 0) kout[row] = bi + 1;
}

// One block per row: binary radix-select of the k-th largest (ordered uint
// keys), exact stable tie handling via index-order prefix scan.
__global__ __launch_bounds__(256) void topk_mask(const float* __restrict__ X,
                                                 const int* __restrict__ kin,
                                                 bf16* __restrict__ khot, int ncols) {
    __shared__ unsigned keys[1024];
    __shared__ unsigned tsum[256];
    __shared__ unsigned cnt;
    const int row = blockIdx.x, t = threadIdx.x;
    const float* xr = X + (size_t)row * ncols;
    for (int j = t; j < ncols; j += 256) {
        unsigned u = __float_as_uint(xr[j]);
        keys[j] = (u & 0x80000000u) ? ~u : (u | 0x80000000u);  // order-preserving
    }
    int kwant = kin[row];
    __syncthreads();

    unsigned prefix = 0;
    for (int b = 31; b >= 0; --b) {
        if (t == 0) cnt = 0;
        __syncthreads();
        const unsigned want = (prefix >> b) | 1u;
        unsigned local = 0;
        for (int j = t; j < ncols; j += 256) local += ((keys[j] >> b) == want);
        if (local) atomicAdd(&cnt, local);
        __syncthreads();
        const unsigned c = cnt;
        if (c >= (unsigned)kwant) prefix |= (1u << b);
        else                      kwant -= c;
        __syncthreads();
    }
    // prefix == k-th largest key; kwant == #ties to keep (lowest indices first)
    unsigned eq[4], gt[4], s = 0;
#pragma unroll
    for (int q = 0; q < 4; ++q) {
        const unsigned key = keys[4 * t + q];
        eq[q] = (key == prefix); gt[q] = (key > prefix); s += eq[q];
    }
    tsum[t] = s;
    __syncthreads();
    for (int off = 1; off < 256; off <<= 1) {     // Hillis-Steele inclusive scan
        const unsigned v = (t >= off) ? tsum[t - off] : 0u;
        __syncthreads();
        tsum[t] += v;
        __syncthreads();
    }
    unsigned run = tsum[t] - s;                   // exclusive rank among ties
    bf16* orow = khot + (size_t)row * ncols;
#pragma unroll
    for (int q = 0; q < 4; ++q) {
        const bool sel = gt[q] || (eq[q] && run < (unsigned)kwant);
        run += eq[q];
        orow[4 * t + q] = sel ? (bf16)1.0f : (bf16)0.0f;
    }
}

// ---------------------------------------------------------------------------
extern "C" void kernel_launch(void* const* d_in, const int* in_sizes, int n_in,
                              void* d_out, int out_size, void* d_ws, size_t ws_size,
                              hipStream_t stream) {
    (void)in_sizes; (void)n_in; (void)out_size; (void)ws_size;
    const int B = 32768, Q = 1024, E = 1024;
    const float* x  = (const float*)d_in[0];
    const float* W1 = (const float*)d_in[1];
    const float* W2 = (const float*)d_in[2];
    const float* W3 = (const float*)d_in[3];
    const float* Wc = (const float*)d_in[4];
    float* out = (float*)d_out;

    char* ws = (char*)d_ws;
    const size_t MB = 1024ull * 1024ull;
    bf16*  w1b  = (bf16*)(ws + 0 * MB);     //   4 MB  [2Q, Q]
    bf16*  w2b  = (bf16*)(ws + 4 * MB);     //   4 MB  [Q, 2Q]
    bf16*  w3b  = (bf16*)(ws + 8 * MB);     //   2 MB  [Q, Q]
    bf16*  wcb  = (bf16*)(ws + 10 * MB);    //   2 MB  [E, Q]
    int*   karr = (int*) (ws + 12 * MB);    // 128 KB  [B]
    bf16*  xb   = (bf16*)(ws + 13 * MB);    //  64 MB  [B, Q]
    bf16*  h1b  = (bf16*)(ws + 77 * MB);    // 128 MB  [B, 2Q]
    float* sc   = (float*)(ws + 77 * MB);   // 128 MB  [B, Q]  (aliases dead h1b)
    bf16*  h2b  = (bf16*)(ws + 205 * MB);   //  64 MB  [B, Q]
    bf16*  khot = (bf16*)(ws + 205 * MB);   //  64 MB  [B, Q]  (aliases dead h2b)

    cvt_f32_to_bf16<<<8192, 256, 0, stream>>>(x,  xb,  B * Q);
    cvt_f32_to_bf16<<<4096, 256, 0, stream>>>(W1, w1b, 2 * Q * Q);
    cvt_f32_to_bf16<<<4096, 256, 0, stream>>>(W2, w2b, 2 * Q * Q);
    cvt_f32_to_bf16<<<2048, 256, 0, stream>>>(W3, w3b, Q * Q);
    cvt_f32_to_bf16<<<2048, 256, 0, stream>>>(Wc, wcb, Q * Q);

    const dim3 blk(256);
    // h1 = relu(x @ W1^T)
    gemm_wmma<true,  false><<<dim3(B / 128, (2 * Q) / 128), blk, 0, stream>>>(
        xb, w1b, h1b, 2 * Q, Q);
    // h2 = relu(h1 @ W2^T)
    gemm_wmma<true,  false><<<dim3(B / 128, Q / 128), blk, 0, stream>>>(
        h1b, w2b, h2b, Q, 2 * Q);
    // scores = h2 @ W3^T  (f32, into region of now-dead h1b)
    gemm_wmma<false, true ><<<dim3(B / 128, Q / 128), blk, 0, stream>>>(
        h2b, w3b, sc, Q, Q);
    // k per row, then dynamic top-k mask of x (khot into region of dead h2b)
    argmax_k <<<B / 8, 256, 0, stream>>>(sc, karr, Q);
    topk_mask<<<B,     256, 0, stream>>>(x, karr, khot, Q);
    // out = khot @ Wc^T  (f32 output)
    gemm_wmma<false, true ><<<dim3(B / 128, E / 128), blk, 0, stream>>>(
        khot, wcb, out, E, Q);
}
